// LookupFFN_69965017252061
// MI455X (gfx1250) — matrix operand order repaired and checked by reading, not compiled, not run on hardware
//
#include <hip/hip_runtime.h>

// ---------------------------------------------------------------------------
// LookupFFN for MI455X (gfx1250, wave32, WMMA)
//
// Pipeline:
//   1) proj_w  -> bf16                          (k_cvt_bf16)
//   2) x -> LayerNorm -> bf16                   (k_layernorm_bf16, wave/token)
//   3) scores = xn @ Wb^T + pb  via v_wmma_f32_16x16x32_bf16 (k_proj_wmma)
//   4) per (token,table): 256 factorized logits -> top8 -> softmax
//      (k_topk_softmax, 1 lane per (n,t), nibble-table trick)
//   5) out = tables_bias + sum w * tables_weight[t, idx, :]  (k_combine)
//      tables_weight (75.5MB) is L2-resident on MI455X (192MB L2).
// ---------------------------------------------------------------------------

typedef __bf16 v16bf __attribute__((ext_vector_type(16)));
typedef __bf16 v8bf  __attribute__((ext_vector_type(8)));
typedef float  v8f   __attribute__((ext_vector_type(8)));

static constexpr int HDIM      = 768;
static constexpr int NUM_TABLE = 96;
static constexpr int TABLE_SZ  = 256;
static constexpr int K_QUERY   = 8;
static constexpr int NTK       = NUM_TABLE * K_QUERY;   // 768

// ---------------------------------------------------------------- kernel 1 --
__global__ void k_cvt_bf16(const float* __restrict__ src,
                           __bf16* __restrict__ dst, int n) {
  int i = blockIdx.x * blockDim.x + threadIdx.x;
  if (i < n) dst[i] = (__bf16)src[i];
}

// ---------------------------------------------------------------- kernel 2 --
// One wave (32 lanes) per token; 24 elements per lane; shuffle reductions.
__global__ __launch_bounds__(256)
void k_layernorm_bf16(const float* __restrict__ x,
                      const float* __restrict__ w,
                      const float* __restrict__ b,
                      __bf16* __restrict__ xn, int ntok) {
  const int lane  = threadIdx.x & 31;
  const int wave  = threadIdx.x >> 5;
  const int token = blockIdx.x * 8 + wave;
  if (token >= ntok) return;

  const float* row = x + (size_t)token * HDIM;
  float v[24];
  float s = 0.f;
#pragma unroll
  for (int j = 0; j < 24; ++j) { v[j] = row[lane + 32 * j]; s += v[j]; }
#pragma unroll
  for (int m = 16; m >= 1; m >>= 1) s += __shfl_xor(s, m, 32);
  const float mu = s * (1.0f / HDIM);

  float q = 0.f;
#pragma unroll
  for (int j = 0; j < 24; ++j) { float d = v[j] - mu; q += d * d; }
#pragma unroll
  for (int m = 16; m >= 1; m >>= 1) q += __shfl_xor(q, m, 32);
  const float rstd = rsqrtf(q * (1.0f / HDIM) + 1e-12f);

  __bf16* orow = xn + (size_t)token * HDIM;
#pragma unroll
  for (int j = 0; j < 24; ++j) {
    const int c = lane + 32 * j;
    orow[c] = (__bf16)((v[j] - mu) * rstd * w[c] + b[c]);
  }
}

// ---------------------------------------------------------------- kernel 3 --
// scores[m,n] = sum_k xn[m,k] * Wb[n,k] + pb[n]
// Block = 128 threads (4 waves). Wave tile: 32(M) x 64(N) = 2x4 WMMA C tiles.
// K loop: 24 iterations of K=32 with v_wmma_f32_16x16x32_bf16.
//
// Fragment layouts used by this kernel (wave32; A/B in contiguous K order):
//   A 16x32 bf16 : lane m = lane&15; elems 0..15 = K(16*lh .. 16*lh+15)
//   B 32x16 bf16 : lane col = lane&15; elems 0..15 = K(16*lh .. 16*lh+15)
//   C/D 16x16 f32: lane col = lane&15; vgpr r = row (r + 8*lh)
__global__ __launch_bounds__(128)
void k_proj_wmma(const __bf16* __restrict__ xn,
                 const __bf16* __restrict__ wb,
                 const float* __restrict__ pb,
                 float* __restrict__ scores) {
  const int lane = threadIdx.x & 31;
  const int wave = threadIdx.x >> 5;
  const int lm   = lane & 15;
  const int lh   = lane >> 4;
  const int m0   = blockIdx.x * 128 + wave * 32;
  const int n0   = blockIdx.y * 64;

  const v8f vzero = {};
  v8f acc[2][4];
#pragma unroll
  for (int tm = 0; tm < 2; ++tm)
#pragma unroll
    for (int tn = 0; tn < 4; ++tn) acc[tm][tn] = vzero;

  for (int k0 = 0; k0 < HDIM; k0 += 32) {
    v16bf afrag[2];
#pragma unroll
    for (int tm = 0; tm < 2; ++tm) {
      const __bf16* ap = xn + (size_t)(m0 + tm * 16 + lm) * HDIM + k0 + 16 * lh;
      v16bf a = *(const v16bf*)ap;          // K = k0+16*lh .. +15
      afrag[tm] = a;
    }
    v16bf bfrag[4];
#pragma unroll
    for (int tn = 0; tn < 4; ++tn) {
      const __bf16* bp = wb + (size_t)(n0 + tn * 16 + lm) * HDIM + k0 + 16 * lh;
      bfrag[tn] = *(const v16bf*)bp;        // 16 contiguous K values
    }
#pragma unroll
    for (int tm = 0; tm < 2; ++tm)
#pragma unroll
      for (int tn = 0; tn < 4; ++tn)
        acc[tm][tn] = __builtin_amdgcn_wmma_f32_16x16x32_bf16(
            false, afrag[tm], false, bfrag[tn],
            (short)0, acc[tm][tn], false, false);
  }

#pragma unroll
  for (int tm = 0; tm < 2; ++tm)
#pragma unroll
    for (int tn = 0; tn < 4; ++tn) {
      const int col  = n0 + tn * 16 + lm;
      const float bs = pb[col];
#pragma unroll
      for (int r = 0; r < 8; ++r) {
        const int row = m0 + tm * 16 + 8 * lh + r;
        scores[(size_t)row * HDIM + col] = acc[tm][tn][r] + bs;
      }
    }
}

// ---------------------------------------------------------------- kernel 4 --
// One lane per (token, table). logit(c) = lo[c&15] + hi[c>>4] with
// lo/hi nibble sums of +-scores. Sorted 8-deep insertion (strict >, so ties
// keep the lower index, matching lax.top_k), then fused softmax.
__global__ __launch_bounds__(256)
void k_topk_softmax(const float* __restrict__ scores,
                    float* __restrict__ topw,
                    int* __restrict__ topi, int ntt) {
  const int gid = blockIdx.x * blockDim.x + threadIdx.x;
  if (gid >= ntt) return;
  const int n = gid / NUM_TABLE;
  const int t = gid - n * NUM_TABLE;
  const float* s = scores + (size_t)n * HDIM + t * 8;
  const float s0 = s[0], s1 = s[1], s2 = s[2], s3 = s[3];
  const float s4 = s[4], s5 = s[5], s6 = s[6], s7 = s[7];

  float lo[16];
#pragma unroll
  for (int cl = 0; cl < 16; ++cl) {
    lo[cl] = ((cl & 1) ? s0 : -s0) + ((cl & 2) ? s1 : -s1) +
             ((cl & 4) ? s2 : -s2) + ((cl & 8) ? s3 : -s3);
  }

  float v0 = -3.0e38f, v1 = -3.0e38f, v2 = -3.0e38f, v3 = -3.0e38f;
  float v4 = -3.0e38f, v5 = -3.0e38f, v6 = -3.0e38f, v7 = -3.0e38f;
  int   i0 = 0, i1 = 0, i2 = 0, i3 = 0, i4 = 0, i5 = 0, i6 = 0, i7 = 0;

#define INS(vv, ii)                                              \
  { bool c = tv > (vv);                                          \
    float fv = c ? (vv) : tv; int fi = c ? (ii) : ti;            \
    (vv) = c ? tv : (vv);     (ii) = c ? ti : (ii);              \
    tv = fv; ti = fi; }

  for (int ch = 0; ch < 16; ++ch) {
    const float hs = ((ch & 1) ? s4 : -s4) + ((ch & 2) ? s5 : -s5) +
                     ((ch & 4) ? s6 : -s6) + ((ch & 8) ? s7 : -s7);
#pragma unroll
    for (int cl = 0; cl < 16; ++cl) {
      const float v = hs + lo[cl];
      if (v > v7) {
        float tv = v; int ti = (ch << 4) | cl;
        INS(v0, i0) INS(v1, i1) INS(v2, i2) INS(v3, i3)
        INS(v4, i4) INS(v5, i5) INS(v6, i6) INS(v7, i7)
      }
    }
  }
#undef INS

  // softmax over sorted-descending top-8 (v0 is max)
  const float w0 = 1.0f;
  const float w1 = __expf(v1 - v0), w2 = __expf(v2 - v0), w3 = __expf(v3 - v0);
  const float w4 = __expf(v4 - v0), w5 = __expf(v5 - v0), w6 = __expf(v6 - v0);
  const float w7 = __expf(v7 - v0);
  const float inv = 1.0f / (w0 + w1 + w2 + w3 + w4 + w5 + w6 + w7);

  float* ow = topw + (size_t)gid * 8;
  int*   oi = topi + (size_t)gid * 8;
  ow[0] = w0 * inv; ow[1] = w1 * inv; ow[2] = w2 * inv; ow[3] = w3 * inv;
  ow[4] = w4 * inv; ow[5] = w5 * inv; ow[6] = w6 * inv; ow[7] = w7 * inv;
  oi[0] = i0; oi[1] = i1; oi[2] = i2; oi[3] = i3;
  oi[4] = i4; oi[5] = i5; oi[6] = i6; oi[7] = i7;
}

// ---------------------------------------------------------------- kernel 5 --
// One block per token; 3 coalesced output columns per thread; top-8 meta in
// LDS; gathered rows stream from the 192MB L2 (tables_weight = 75.5MB).
__global__ __launch_bounds__(256)
void k_combine(const float* __restrict__ topw,
               const int* __restrict__ topi,
               const float* __restrict__ tw,
               const float* __restrict__ tbias,
               float* __restrict__ out) {
  __shared__ float sw[NTK];
  __shared__ int   si[NTK];
  const int token = blockIdx.x;
  const int tid   = threadIdx.x;

#pragma unroll
  for (int j = 0; j < 3; ++j) {
    const int idx = tid + 256 * j;
    sw[idx] = topw[(size_t)token * NTK + idx];
    si[idx] = topi[(size_t)token * NTK + idx];
  }
  __syncthreads();

  float a0 = tbias[tid];
  float a1 = tbias[tid + 256];
  float a2 = tbias[tid + 512];

#pragma unroll 4
  for (int r = 0; r < NTK; ++r) {
    const float w  = sw[r];
    const int   ix = si[r];
    const int   t  = r >> 3;
    const float* row = tw + ((size_t)(t * TABLE_SZ + ix)) * HDIM;
    if (r + 8 < NTK) {
      const int t2 = (r + 8) >> 3;
      __builtin_prefetch(tw + ((size_t)(t2 * TABLE_SZ + si[r + 8])) * HDIM + tid, 0, 0);
    }
    a0 = fmaf(w, row[tid], a0);
    a1 = fmaf(w, row[tid + 256], a1);
    a2 = fmaf(w, row[tid + 512], a2);
  }

  float* orow = out + (size_t)token * HDIM;
  orow[tid] = a0; orow[tid + 256] = a1; orow[tid + 512] = a2;
}

// ------------------------------------------------------------------- host --
extern "C" void kernel_launch(void* const* d_in, const int* in_sizes, int n_in,
                              void* d_out, int out_size, void* d_ws, size_t ws_size,
                              hipStream_t stream) {
  const float* x   = (const float*)d_in[0];   // [N, H]
  const float* lnw = (const float*)d_in[1];   // [H]
  const float* lnb = (const float*)d_in[2];   // [H]
  const float* pw  = (const float*)d_in[3];   // [H, H]
  const float* pb  = (const float*)d_in[4];   // [H]
  const float* tw  = (const float*)d_in[5];   // [96, 256, H]
  const float* tb  = (const float*)d_in[6];   // [H]
  float* out = (float*)d_out;

  const int H    = in_sizes[1];               // 768
  const int ntok = in_sizes[0] / H;           // 4096

  char* ws = (char*)d_ws;
  size_t off = 0;
  auto align256 = [](size_t v) { return (v + 255) & ~(size_t)255; };

  __bf16* wb = (__bf16*)(ws + off); off = align256(off + (size_t)H * H * 2);
  __bf16* xn = (__bf16*)(ws + off); off = align256(off + (size_t)ntok * H * 2);
  float* scores = (float*)(ws + off); off = align256(off + (size_t)ntok * H * 4);
  float* topw   = (float*)(ws + off); off = align256(off + (size_t)ntok * NTK * 4);
  int*   topi   = (int*)(ws + off);   off = align256(off + (size_t)ntok * NTK * 4);
  (void)ws_size; (void)n_in; (void)out_size;

  // 1) proj_w -> bf16
  k_cvt_bf16<<<(H * H + 255) / 256, 256, 0, stream>>>(pw, wb, H * H);
  // 2) LayerNorm -> bf16 activations (8 tokens per 256-thread block)
  k_layernorm_bf16<<<(ntok + 7) / 8, 256, 0, stream>>>(x, lnw, lnb, xn, ntok);
  // 3) WMMA projection GEMM: [ntok, H] x [H, H]^T
  dim3 gg(ntok / 128, H / 64);
  k_proj_wmma<<<gg, 128, 0, stream>>>(xn, wb, pb, scores);
  // 4) factorized logits -> top-8 -> softmax
  const int ntt = ntok * NUM_TABLE;
  k_topk_softmax<<<(ntt + 255) / 256, 256, 0, stream>>>(scores, topw, topi, ntt);
  // 5) gather-combine from L2-resident tables
  k_combine<<<ntok, 256, 0, stream>>>(topw, topi, tw, tb, out);
}
